// WordsLstm_9277129359702
// MI455X (gfx1250) — matrix operand order repaired and checked by reading, not compiled, not run on hardware
//
#include <hip/hip_runtime.h>
#include <hip/hip_bf16.h>
#include <stdint.h>

// Problem dims (fixed by reference)
#define B_  64
#define T_  512
#define E_  512
#define H_  512
#define NWG 32           // H_/16 column chunks
#define LSTM_THREADS 128 // 4 waves (wave32)

typedef __attribute__((ext_vector_type(16))) __bf16 v16bf;
typedef __attribute__((ext_vector_type(8)))  float  v8f;

union FragU { uint4 u[2]; v16bf v; };

// ---------------- prologue kernels ----------------

__global__ void embed_gather_kernel(const int* __restrict__ x,
                                    const float* __restrict__ emb,
                                    __hip_bfloat16* __restrict__ xs) {
  const int row = blockIdx.x;              // row = b*T + t
  const int tok = x[row];
  const float* src = emb + (size_t)tok * E_;
  __hip_bfloat16* dst = xs + (size_t)row * E_;
  const int e = threadIdx.x * 4;           // 128 threads x 4 = 512
  const float4 f = *reinterpret_cast<const float4*>(src + e);
  union { __hip_bfloat16 h[4]; uint2 u; } p;
  p.h[0] = __float2bfloat16(f.x);
  p.h[1] = __float2bfloat16(f.y);
  p.h[2] = __float2bfloat16(f.z);
  p.h[3] = __float2bfloat16(f.w);
  *reinterpret_cast<uint2*>(dst + e) = p.u;
}

__global__ void init_h_kernel(const float* __restrict__ h0,
                              __hip_bfloat16* __restrict__ hbuf) {
  int i = blockIdx.x * blockDim.x + threadIdx.x;
  if (i < B_ * H_) hbuf[i] = __float2bfloat16(h0[i]);
}

// ---------------- activation helpers (hardware v_tanh_f32) ----------------

__device__ __forceinline__ float fast_tanh(float x) {
#if __has_builtin(__builtin_amdgcn_tanhf)
  return __builtin_amdgcn_tanhf(x);
#else
  return tanhf(x);
#endif
}
__device__ __forceinline__ float fast_sigmoid(float x) {
  return 0.5f + 0.5f * fast_tanh(0.5f * x);
}

// ---------------- 16-block WMMA phase (K = 512) ----------------
// LDS layout: sW[kb][gate][n][32k] bf16 -> one kb-block = 4KB, all gate/frag
// offsets are DS-immediate-encodable from a single per-phase base.
__device__ __forceinline__ void mm_phase(v8f acc[4],
                                         const __hip_bfloat16* __restrict__ arow,
                                         const __hip_bfloat16* __restrict__ wlane) {
#pragma unroll
  for (int kb = 0; kb < 16; ++kb) {
    FragU a;
    const __hip_bfloat16* ap = arow + kb * 32;
    a.u[0] = *reinterpret_cast<const uint4*>(ap);
    a.u[1] = *reinterpret_cast<const uint4*>(ap + 16);

    const __hip_bfloat16* wp = wlane + kb * 2048;   // 4KB stride per K-block
    FragU b0, b1, b2, b3;                            // all 4 gates up front
    b0.u[0] = *reinterpret_cast<const uint4*>(wp);
    b0.u[1] = *reinterpret_cast<const uint4*>(wp + 16);
    b1.u[0] = *reinterpret_cast<const uint4*>(wp + 512);
    b1.u[1] = *reinterpret_cast<const uint4*>(wp + 528);
    b2.u[0] = *reinterpret_cast<const uint4*>(wp + 1024);
    b2.u[1] = *reinterpret_cast<const uint4*>(wp + 1040);
    b3.u[0] = *reinterpret_cast<const uint4*>(wp + 1536);
    b3.u[1] = *reinterpret_cast<const uint4*>(wp + 1552);

    acc[0] = __builtin_amdgcn_wmma_f32_16x16x32_bf16(false, a.v, false, b0.v, (short)0, acc[0], false, false);
    acc[1] = __builtin_amdgcn_wmma_f32_16x16x32_bf16(false, a.v, false, b1.v, (short)0, acc[1], false, false);
    acc[2] = __builtin_amdgcn_wmma_f32_16x16x32_bf16(false, a.v, false, b2.v, (short)0, acc[2], false, false);
    acc[3] = __builtin_amdgcn_wmma_f32_16x16x32_bf16(false, a.v, false, b3.v, (short)0, acc[3], false, false);
  }
}

// ---------------- persistent recurrent kernel ----------------

__global__ void __launch_bounds__(LSTM_THREADS, 1)
lstm_persistent_kernel(const __hip_bfloat16* __restrict__ xs,   // [B*T, E] bf16
                       __hip_bfloat16* __restrict__ hbuf,       // [2][B, H] bf16
                       const float* __restrict__ W_ih,          // [4H, E]
                       const float* __restrict__ W_hh,          // [4H, H]
                       const float* __restrict__ b_ih,          // [4H]
                       const float* __restrict__ b_hh,          // [4H]
                       const float* __restrict__ c0,            // [B, H]
                       float* __restrict__ out,                 // [B, T, H]
                       unsigned* __restrict__ counter) {
  extern __shared__ char smem[];
  __hip_bfloat16* sW = reinterpret_cast<__hip_bfloat16*>(smem); // [32 kb][4 g][16 n][32 k] = 128 KB

  const int tid  = threadIdx.x;
  const int wave = tid >> 5;
  const int lane = tid & 31;
  const int lm   = lane & 15;
  const int lh   = lane >> 4;
  const int col0 = blockIdx.x * 16;

  // ---- one-time preload: fused [W_ih | W_hh] slice -> LDS (bf16) ----
  // sW[kb*2048 + g*512 + n*32 + kk] = Wcat[g*H + col0 + n][kb*32 + kk]
  for (int idx = tid; idx < 4 * 16 * 1024; idx += LSTM_THREADS) {
    const int kb = idx >> 11;
    const int g  = (idx >> 9) & 3;
    const int n  = (idx >> 5) & 15;
    const int k  = (kb << 5) | (idx & 31);
    const int grow = g * H_ + col0 + n;
    const float w = (k < E_) ? W_ih[(size_t)grow * E_ + k]
                             : W_hh[(size_t)grow * H_ + (k - E_)];
    sW[idx] = __float2bfloat16(w);
  }
  __syncthreads();

  // ---- per-lane persistent state ----
  const int col = col0 + lm;
  const int bB  = wave * 16;
  float bias[4];
#pragma unroll
  for (int g = 0; g < 4; ++g)
    bias[g] = b_ih[g * H_ + col] + b_hh[g * H_ + col];

  float c_reg[8];
#pragma unroll
  for (int r = 0; r < 8; ++r)
    c_reg[r] = c0[(size_t)(bB + r + 8 * lh) * H_ + col];

  const __hip_bfloat16* xrow0 = xs + (size_t)(bB + lm) * T_ * E_ + lh * 8;
  const __hip_bfloat16* hrow0 = hbuf + (size_t)(bB + lm) * H_ + lh * 8;
  const __hip_bfloat16* wlane  = sW + lm * 32 + lh * 8;        // phase-1 base (kb 0..15)
  const __hip_bfloat16* wlane2 = wlane + 16 * 2048;            // phase-2 base (kb 16..31)

  const v8f vzero = {0.f, 0.f, 0.f, 0.f, 0.f, 0.f, 0.f, 0.f};

  v8f acc[4];
#pragma unroll
  for (int g = 0; g < 4; ++g) acc[g] = vzero;
  mm_phase(acc, xrow0, wlane);                                 // phase-1 for t = 0

  for (int t = 0; t < T_; ++t) {
    if (t + 1 < T_)                                            // lead prefetch for next phase-1
      __builtin_prefetch(xrow0 + (size_t)(t + 1) * E_, 0, 1);

    // ---- wait until h_t (written at step t-1) is globally visible ----
    __syncthreads();
    if (tid == 0) {
      const unsigned target = (unsigned)(NWG * t);
      while (__hip_atomic_load(counter, __ATOMIC_ACQUIRE,
                               __HIP_MEMORY_SCOPE_AGENT) < target)
        __builtin_amdgcn_s_sleep(2);
    }
    __syncthreads();
    __threadfence();                                           // acquire

    // ---- phase 2: += h_t @ W_hh^T ----
    mm_phase(acc, hrow0 + (size_t)(t & 1) * (B_ * H_), wlane2);

    // ---- LSTM cell epilogue (hw tanh; cell state in registers) ----
    __hip_bfloat16* hnxt = (__hip_bfloat16*)(hrow0 - (size_t)lh * 8) -
                           (size_t)(bB + lm) * H_;             // == hbuf
    hnxt = (__hip_bfloat16*)hbuf + (size_t)((t + 1) & 1) * (B_ * H_);
#pragma unroll
    for (int r = 0; r < 8; ++r) {
      const int b = bB + r + 8 * lh;
      const float iv = fast_sigmoid(acc[0][r] + bias[0]);
      const float fv = fast_sigmoid(acc[1][r] + bias[1]);
      const float gv = fast_tanh  (acc[2][r] + bias[2]);
      const float ov = fast_sigmoid(acc[3][r] + bias[3]);
      const float c  = fv * c_reg[r] + iv * gv;
      c_reg[r] = c;
      const float h = ov * fast_tanh(c);
      out[((size_t)b * T_ + t) * H_ + col] = h;
      hnxt[(size_t)b * H_ + col] = __float2bfloat16(h);
    }

    // ---- split barrier: signal now, hide latency under next phase-1 ----
    __syncthreads();
    if (tid == 0) {
      __threadfence();                                         // release h writes
      atomicAdd(counter, 1u);
    }

    if (t + 1 < T_) {
#pragma unroll
      for (int g = 0; g < 4; ++g) acc[g] = vzero;
      mm_phase(acc, xrow0 + (size_t)(t + 1) * E_, wlane);      // phase-1 for t+1
    }
  }
}

// ---------------- host launch ----------------

extern "C" void kernel_launch(void* const* d_in, const int* in_sizes, int n_in,
                              void* d_out, int out_size, void* d_ws, size_t ws_size,
                              hipStream_t stream) {
  (void)in_sizes; (void)n_in; (void)out_size; (void)ws_size;
  const int*   x    = (const int*)  d_in[0];
  const float* h0   = (const float*)d_in[1];
  const float* c0   = (const float*)d_in[2];
  const float* emb  = (const float*)d_in[3];
  const float* W_ih = (const float*)d_in[4];
  const float* W_hh = (const float*)d_in[5];
  const float* b_ih = (const float*)d_in[6];
  const float* b_hh = (const float*)d_in[7];
  float* out = (float*)d_out;

  uint8_t* ws = (uint8_t*)d_ws;
  unsigned*       counter = (unsigned*)ws;                                // 64 B
  __hip_bfloat16* hbuf    = (__hip_bfloat16*)(ws + 64);                   // 2*B*H*2 = 128 KB
  __hip_bfloat16* xs      = (__hip_bfloat16*)(ws + 64 + 2 * B_ * H_ * 2); // B*T*E*2 = 32 MB

  hipMemsetAsync(counter, 0, 64, stream);
  init_h_kernel<<<(B_ * H_ + 255) / 256, 256, 0, stream>>>(h0, hbuf);
  embed_gather_kernel<<<B_ * T_, 128, 0, stream>>>(x, emb, xs);

  const int lds_bytes = 4 * 16 * 1024 * (int)sizeof(__hip_bfloat16);      // 128 KB of 320 KB LDS
  (void)hipFuncSetAttribute(reinterpret_cast<const void*>(lstm_persistent_kernel),
                            hipFuncAttributeMaxDynamicSharedMemorySize, lds_bytes);
  lstm_persistent_kernel<<<NWG, LSTM_THREADS, lds_bytes, stream>>>(
      xs, hbuf, W_ih, W_hh, b_ih, b_hh, c0, out, counter);
}
